// WindowAttention_Kai_32169305047715
// MI455X (gfx1250) — matrix operand order repaired and checked by reading, not compiled, not run on hardware
//
#include <hip/hip_runtime.h>

typedef __attribute__((ext_vector_type(16))) _Float16 v16h;
typedef __attribute__((ext_vector_type(8)))  float    v8f;
typedef unsigned int u32x4 __attribute__((ext_vector_type(4)));
typedef int          i32x4 __attribute__((ext_vector_type(4)));
typedef int          i32x8 __attribute__((ext_vector_type(8)));

#define WSZ    8
#define NTOK   64          // tokens per window
#define CCH    192         // d_model
#define NHEADS 6
#define DHEAD  32
#define C3     576         // 3*C
#define NBIAS  225         // (2*8-1)^2

// ---------------- WMMA fragment helpers (wave32, contiguous k = 16*half + element; not the ISA table order) ---------------------

// A matrix 16x32 f16: lanes 0-15 -> M=0..15, elements 0..15 = K 0..15
//                     lanes 16-31 -> M=0..15, elements 0..15 = K 16..31
__device__ __forceinline__ v16h load_A_lds(const _Float16* __restrict__ base, int ld, int lane) {
  const int m = lane & 15, hf = lane >> 4;
  const _Float16* p = base + m * ld;
  v16h a;
#pragma unroll
  for (int j = 0; j < 16; ++j) a[j] = p[hf * 16 + j];
  return a;
}

// B matrix 32x16 f16, B(k,n) at base[n*ld + k] (contiguous in k) from LDS.
__device__ __forceinline__ v16h load_Bt_lds(const _Float16* __restrict__ base, int ld, int lane) {
  const int n = lane & 15, hf = lane >> 4;
  const _Float16* p = base + n * ld + hf * 16;
  v16h b;
#pragma unroll
  for (int j = 0; j < 16; ++j) b[j] = p[j];
  return b;
}

// B matrix 32x16 f16, B(k,n) at base[k*ld + n] (row-major K x N) from LDS.
__device__ __forceinline__ v16h load_B_lds(const _Float16* __restrict__ base, int ld, int lane) {
  const int n = lane & 15, hf = lane >> 4;
  v16h b;
#pragma unroll
  for (int j = 0; j < 16; ++j) b[j] = base[(hf * 16 + j) * ld + n];
  return b;
}

// B matrix 32x16 from global weight in W^T layout: B(k,n) = w[off + n*ld + k] (contig in k).
// F16W: weights pre-converted to f16 in scratch; else raw f32 with on-the-fly convert.
template <bool F16W>
__device__ __forceinline__ v16h load_Bt_w(const void* __restrict__ w, size_t off, int ld, int lane) {
  const int n = lane & 15, hf = lane >> 4;
  v16h b;
  if constexpr (F16W) {
    const _Float16* p = (const _Float16*)w + off + (size_t)n * ld + hf * 16;
#pragma unroll
    for (int j = 0; j < 16; ++j) b[j] = p[j];
  } else {
    const float* p = (const float*)w + off + (size_t)n * ld + hf * 16;
#pragma unroll
    for (int j = 0; j < 16; ++j) b[j] = (_Float16)p[j];
  }
  return b;
}

// ---------------- TDM: async 2D tile load (window of x) global -> LDS --------------------
// Descriptor per CDNA5 ISA 8.3/8.4. 2D tensor: dim0 = W*C elements/row, dim1 = H rows.
// Tile: tile_dim0 = 8*192 = 1536 contiguous f32 (one window row), tile_dim1 = 8 rows.
__device__ __forceinline__ void tdm_load_window(unsigned lds_off, const float* gptr,
                                                int Ww, int Hh) {
  const unsigned long long ga = (unsigned long long)(uintptr_t)gptr;
  u32x4 g0;
  g0[0] = 1u;                                        // count=1, user descriptor
  g0[1] = lds_off;                                   // lds_addr (bytes)
  g0[2] = (unsigned)ga;                              // global_addr[31:0]
  g0[3] = (unsigned)(ga >> 32) | 0x80000000u;        // global_addr[56:32] | type=2

  const unsigned td0 = (unsigned)Ww * CCH;           // tensor_dim0 (elements)
  const unsigned td1 = (unsigned)Hh;                 // tensor_dim1 (rows)
  const unsigned long long st0 = (unsigned long long)Ww * CCH;  // dim0 stride
  i32x8 g1;
  g1[0] = 0x20000;                                   // data_size = 2 (4 bytes)
  g1[1] = (int)((td0 & 0xFFFFu) << 16);              // tensor_dim0[15:0]
  g1[2] = (int)((td0 >> 16) | ((td1 & 0xFFFFu) << 16));
  g1[3] = (int)((td1 >> 16) | (((unsigned)(WSZ * CCH)) << 16)); // tile_dim0 = 1536
  g1[4] = WSZ;                                       // tile_dim1 = 8, tile_dim2 = 0
  g1[5] = (int)(unsigned)(st0 & 0xFFFFFFFFu);        // tensor_dim0_stride lo
  g1[6] = (int)(unsigned)(st0 >> 32);                // stride hi | dim1_stride lo (0)
  g1[7] = 0;

  i32x4 z4 = {0, 0, 0, 0};
#if defined(__clang_major__) && (__clang_major__ >= 23)
  i32x8 z8 = {0, 0, 0, 0, 0, 0, 0, 0};
  __builtin_amdgcn_tensor_load_to_lds(g0, g1, z4, z4, z8, 0);
#else
  __builtin_amdgcn_tensor_load_to_lds(g0, g1, z4, z4, 0);
#endif
}

// ---------------- one-shot weight convert f32 -> f16 into scratch ------------------------
__global__ void convert_weights(const float* __restrict__ qkv_w,
                                const float* __restrict__ proj_w,
                                _Float16* __restrict__ wh) {
  const int i = blockIdx.x * blockDim.x + threadIdx.x;
  const int nq = C3 * CCH, np = CCH * CCH;
  if (i < nq)            wh[i] = (_Float16)qkv_w[i];
  else if (i < nq + np)  wh[i] = (_Float16)proj_w[i - nq];
}

// ---------------- fused window-attention kernel: 1 block = 1 window, 6 wave32 ------------

template <bool F16W>
__global__ __launch_bounds__(192)
void swin_window_attn(const float* __restrict__ x,
                      const void*  __restrict__ qkvw,
                      const float* __restrict__ qkv_b,
                      const void*  __restrict__ projw,
                      const float* __restrict__ proj_b,
                      const float* __restrict__ rpb,
                      const int*   __restrict__ Hp,
                      const int*   __restrict__ Wp,
                      float* __restrict__ out) {
  // LDS: ~116.5 KB of the 320 KB WGP pool (f32 x-tile overlays s_qkv, dead by phase 1).
  __shared__ _Float16 s_qkv[NTOK * C3];            // 72 KB : QKV activations (f16)
  __shared__ _Float16 s_xo[NTOK * CCH];            // 24 KB : x (f16), later O
  __shared__ _Float16 s_p[NHEADS * 16 * NTOK];     // 12 KB : per-wave softmax(P) row-block
  __shared__ float    s_rpb[NBIAS * NHEADS];       // 5.4 KB
  __shared__ float    s_qkvb[C3];
  __shared__ float    s_projb[CCH];

  const int Hh = Hp[0], Ww = Wp[0];
  const int nW  = Ww >> 3;
  const int nHW = (Hh >> 3) * nW;
  const int win = blockIdx.x;
  const int b   = win / nHW;
  const int wr  = win % nHW;
  const int wy  = wr / nW, wx = wr % nW;

  const int tid  = threadIdx.x;
  const int wave = tid >> 5;        // 0..5 (== head id in phase 2)
  const int lane = tid & 31;
  const int nl   = lane & 15, hf = lane >> 4;

  const float* xbase = x + (size_t)b * Hh * Ww * CCH;
  float*       obase = out + (size_t)b * Hh * Ww * CCH;

  // ---- Phase 0: TDM the x window into LDS (f32), stage tables, convert to f16 -----------
  float* s_xf32 = reinterpret_cast<float*>(s_qkv);          // 48 KB overlay
  const float* gwin = xbase + ((size_t)(wy * WSZ) * Ww + wx * WSZ) * CCH;
  if (wave == 0)
    tdm_load_window((unsigned)(uintptr_t)s_xf32, gwin, Ww, Hh);

  // warm L2 with the weight stream while the TDM runs
  __builtin_prefetch((const char*)qkvw + (size_t)tid * 1024, 0, 1);
  __builtin_prefetch((const char*)projw + (size_t)tid * 512, 0, 1);

  for (int i = tid; i < NBIAS * NHEADS; i += blockDim.x) s_rpb[i] = rpb[i];
  for (int i = tid; i < C3; i += blockDim.x)  s_qkvb[i]  = qkv_b[i];
  for (int i = tid; i < CCH; i += blockDim.x) s_projb[i] = proj_b[i];

  if (wave == 0) __builtin_amdgcn_s_wait_tensorcnt((short)0);
  __syncthreads();

  for (int i = tid; i < NTOK * (CCH / 4); i += blockDim.x) {
    const float4 v = *reinterpret_cast<const float4*>(s_xf32 + i * 4);
    s_xo[i * 4 + 0] = (_Float16)v.x;
    s_xo[i * 4 + 1] = (_Float16)v.y;
    s_xo[i * 4 + 2] = (_Float16)v.z;
    s_xo[i * 4 + 3] = (_Float16)v.w;
  }
  __syncthreads();

  const v8f zacc = {};

  // ---- Phase 1: QKV = x @ qkv_w^T + b  (64x192 . 192x576), n-tiles split over 6 waves ---
  for (int nt = wave; nt < C3 / 16; nt += NHEADS) {
    const int n0 = nt * 16;
    v16h bf[6];
#pragma unroll
    for (int kk = 0; kk < 6; ++kk)
      bf[kk] = load_Bt_w<F16W>(qkvw, (size_t)n0 * CCH + kk * 32, CCH, lane);
#pragma unroll
    for (int mi = 0; mi < 4; ++mi) {
      v8f acc = zacc;
#pragma unroll
      for (int kk = 0; kk < 6; ++kk) {
        v16h a = load_A_lds(&s_xo[(mi * 16) * CCH + kk * 32], CCH, lane);
        acc = __builtin_amdgcn_wmma_f32_16x16x32_f16(false, a, false, bf[kk],
                                                     (short)0, acc, false, false);
      }
      const float bias = s_qkvb[n0 + nl];
#pragma unroll
      for (int r = 0; r < 8; ++r)
        s_qkv[(mi * 16 + r + 8 * hf) * C3 + n0 + nl] = (_Float16)(acc[r] + bias);
    }
  }
  __syncthreads();

  // ---- Phase 2: per-head attention (wave == head) ---------------------------------------
  {
    const int hd = wave;
    const float scale = 0.1767766952966369f;      // 1/sqrt(32)
    _Float16* pbuf = &s_p[hd * 16 * NTOK];        // 16 x 64 staging for this wave

#pragma unroll 1
    for (int mi = 0; mi < 4; ++mi) {
      // S row-block: Q(mi) @ K^T -> 4 tiles of 16x16 f32
      v16h aq = load_A_lds(&s_qkv[(mi * 16) * C3 + hd * DHEAD], C3, lane);
      v8f s[4];
#pragma unroll
      for (int ni = 0; ni < 4; ++ni) {
        v16h bk = load_Bt_lds(&s_qkv[(ni * 16) * C3 + CCH + hd * DHEAD], C3, lane);
        s[ni] = __builtin_amdgcn_wmma_f32_16x16x32_f16(false, aq, false, bk,
                                                       (short)0, zacc, false, false);
      }
      // scale + relative-position bias (index computed analytically)
#pragma unroll
      for (int ni = 0; ni < 4; ++ni)
#pragma unroll
        for (int r = 0; r < 8; ++r) {
          const int m = mi * 16 + r + 8 * hf;     // query token
          const int n = ni * 16 + nl;             // key token
          const int rel = (((m >> 3) - (n >> 3)) + 7) * 15 + (((m & 7) - (n & 7)) + 7);
          s[ni][r] = s[ni][r] * scale + s_rpb[rel * NHEADS + hd];
        }
      // softmax across the 64 keys (4 tiles x 16-lane half-group reductions)
#pragma unroll
      for (int r = 0; r < 8; ++r) {
        float mx = fmaxf(fmaxf(s[0][r], s[1][r]), fmaxf(s[2][r], s[3][r]));
#pragma unroll
        for (int off = 1; off < 16; off <<= 1) mx = fmaxf(mx, __shfl_xor(mx, off, 32));
        float sum = 0.f;
#pragma unroll
        for (int ni = 0; ni < 4; ++ni) { float e = __expf(s[ni][r] - mx); s[ni][r] = e; sum += e; }
#pragma unroll
        for (int off = 1; off < 16; off <<= 1) sum += __shfl_xor(sum, off, 32);
        const float inv = 1.0f / sum;
#pragma unroll
        for (int ni = 0; ni < 4; ++ni) s[ni][r] *= inv;
      }
      // stage P row-block as f16 (A-layout source for P@V)
#pragma unroll
      for (int ni = 0; ni < 4; ++ni)
#pragma unroll
        for (int r = 0; r < 8; ++r)
          pbuf[(r + 8 * hf) * NTOK + ni * 16 + nl] = (_Float16)s[ni][r];

      // O(mi) = P(mi) @ V  (16x64 . 64x32)
      v16h a0 = load_A_lds(pbuf, NTOK, lane);
      v16h a1 = load_A_lds(pbuf + 32, NTOK, lane);
#pragma unroll
      for (int ni = 0; ni < 2; ++ni) {
        v16h b0 = load_B_lds(&s_qkv[2 * CCH + hd * DHEAD + ni * 16], C3, lane);
        v16h b1 = load_B_lds(&s_qkv[32 * C3 + 2 * CCH + hd * DHEAD + ni * 16], C3, lane);
        v8f acc = __builtin_amdgcn_wmma_f32_16x16x32_f16(false, a0, false, b0,
                                                         (short)0, zacc, false, false);
        acc = __builtin_amdgcn_wmma_f32_16x16x32_f16(false, a1, false, b1,
                                                     (short)0, acc, false, false);
#pragma unroll
        for (int r = 0; r < 8; ++r)
          s_xo[(mi * 16 + r + 8 * hf) * CCH + hd * DHEAD + ni * 16 + nl] = (_Float16)acc[r];
      }
    }
  }
  __syncthreads();

  // ---- Phase 3: out = O @ proj_w^T + b, window-reverse scatter to global ----------------
  for (int nt = wave; nt < CCH / 16; nt += NHEADS) {
    const int n0 = nt * 16;
    v16h bf[6];
#pragma unroll
    for (int kk = 0; kk < 6; ++kk)
      bf[kk] = load_Bt_w<F16W>(projw, (size_t)n0 * CCH + kk * 32, CCH, lane);
#pragma unroll
    for (int mi = 0; mi < 4; ++mi) {
      v8f acc = zacc;
#pragma unroll
      for (int kk = 0; kk < 6; ++kk) {
        v16h a = load_A_lds(&s_xo[(mi * 16) * CCH + kk * 32], CCH, lane);
        acc = __builtin_amdgcn_wmma_f32_16x16x32_f16(false, a, false, bf[kk],
                                                     (short)0, acc, false, false);
      }
      const float bias = s_projb[n0 + nl];
#pragma unroll
      for (int r = 0; r < 8; ++r) {
        const int t = mi * 16 + r + 8 * hf;
        const int row = (wy * WSZ + (t >> 3)) * Ww + (wx * WSZ + (t & 7));
        obase[(size_t)row * CCH + n0 + nl] = acc[r] + bias;
      }
    }
  }
}

extern "C" void kernel_launch(void* const* d_in, const int* in_sizes, int n_in,
                              void* d_out, int out_size, void* d_ws, size_t ws_size,
                              hipStream_t stream) {
  const float* x      = (const float*)d_in[0];
  const float* qkv_w  = (const float*)d_in[1];
  const float* qkv_b  = (const float*)d_in[2];
  const float* proj_w = (const float*)d_in[3];
  const float* proj_b = (const float*)d_in[4];
  const float* rpb    = (const float*)d_in[5];
  const int*   Hp     = (const int*)d_in[6];
  const int*   Wp     = (const int*)d_in[7];
  float* out = (float*)d_out;

  const int nwin = in_sizes[0] / (CCH * NTOK);   // B*H*W/64 windows
  dim3 grid(nwin), block(192);

  const int    wElems = C3 * CCH + CCH * CCH;    // qkv_w + proj_w elements
  const size_t wsNeed = (size_t)wElems * sizeof(_Float16);

  if (d_ws != nullptr && ws_size >= wsNeed) {
    _Float16* wh = (_Float16*)d_ws;
    convert_weights<<<(wElems + 255) / 256, 256, 0, stream>>>(qkv_w, proj_w, wh);
    swin_window_attn<true><<<grid, block, 0, stream>>>(
        x, (const void*)wh, qkv_b, (const void*)(wh + C3 * CCH), proj_b, rpb, Hp, Wp, out);
  } else {
    swin_window_attn<false><<<grid, block, 0, stream>>>(
        x, (const void*)qkv_w, qkv_b, (const void*)proj_w, proj_b, rpb, Hp, Wp, out);
  }
}
